// CRF_5669356834219
// MI455X (gfx1250) — matrix operand order repaired and checked by reading, not compiled, not run on hardware
//
#include <hip/hip_runtime.h>

// CRF log-likelihood for B=512, T=1024, L=64 on gfx1250 (MI455X).
//
// Denominator uses the *scaled* forward algorithm so each step is a real
// matmul q = expTrans^T @ p, executed with V_WMMA_F32_16X16X32_F16:
//   - one wave32 per 16 batches (32 workgroups)
//   - A = exp(trans)^T kept in registers the whole scan (const operand)
//   - B = normalized probability columns rebuilt per step from LDS
//   - per-batch running log-scale in f32; p renormalized to rowmax=1 each
//     step, so f16 WMMA inputs are well-conditioned
//   - masked steps keep old p via a second LDS array (pT)
//   - h streamed with contiguous b128 loads + global_prefetch ahead

#define TT 1024
#define LL 64
#define BB 512

typedef __attribute__((ext_vector_type(16))) _Float16 v16h;
typedef __attribute__((ext_vector_type(8)))  float    v8f;

// ---------------------------------------------------------------- numerator
__global__ __launch_bounds__(256) void crf_num_kernel(
    const float* __restrict__ h, const long long* __restrict__ labels,
    const unsigned char* __restrict__ mask,
    const float* __restrict__ trans, const float* __restrict__ start_t,
    const float* __restrict__ end_t, float* __restrict__ out)
{
  int b = blockIdx.x;
  int tid = threadIdx.x;
  const long long* lab = labels + (size_t)b * TT;
  const unsigned char* mrow = mask + (size_t)b * TT;
  const float* hrow = h + (size_t)b * TT * LL;

  float acc = 0.f;
  int cnt = 0;
  for (int t = tid; t < TT; t += 256) {
    int mt = mrow[t];
    cnt += mt;
    if (t < TT - 1) {
      int lt = (int)lab[t];
      int ln = (int)lab[t + 1];
      float e  = hrow[(size_t)t * LL + lt];
      float tg = trans[lt * LL + ln];
      acc += (mt ? e : 0.f) + (mrow[t + 1] ? tg : 0.f);
    }
  }

  __shared__ float sacc[256];
  __shared__ int   scnt[256];
  sacc[tid] = acc;
  scnt[tid] = cnt;
  __syncthreads();
  for (int s = 128; s > 0; s >>= 1) {
    if (tid < s) { sacc[tid] += sacc[tid + s]; scnt[tid] += scnt[tid + s]; }
    __syncthreads();
  }
  if (tid == 0) {
    int last_idx = scnt[0] - 1;
    if (last_idx < 0) last_idx = 0;
    int lastLab = (int)lab[last_idx];
    float num = sacc[0];
    num += start_t[(int)lab[0]];
    num += (mrow[TT - 1] ? hrow[(size_t)(TT - 1) * LL + lastLab] : 0.f);
    num += end_t[lastLab];
    out[b] = num;
  }
}

// -------------------------------------------------------------- denominator
__global__ __launch_bounds__(32) void crf_den_kernel(
    const float* __restrict__ h, const unsigned char* __restrict__ mask,
    const float* __restrict__ trans, const float* __restrict__ start_t,
    const float* __restrict__ end_t, float* __restrict__ out)
{
  // qT[n][j]: raw WMMA output q(t) (state j, batch n), stride 68 floats so
  // b128 reloads are bank-conflict free. pT[n][j]: last normalized p.
  __shared__ alignas(16) float qT[16][68];
  __shared__ alignas(16) float pT[16][68];
  __shared__ unsigned char mLDS[TT * 16];   // mask transposed: [t][batch]

  const int  l    = threadIdx.x;
  const int  n    = l & 15;        // batch lane (column of B/D)
  const bool hi   = l >= 16;
  const int  koff = hi ? 16 : 0;   // this lane's K-offset within a 32-block
  const int  b0   = blockIdx.x * 16;

  // ---- transpose mask for this batch group into LDS (one-time)
  {
    const unsigned char* mrow = mask + (size_t)(b0 + n) * TT;
    int tbase = hi ? (TT / 2) : 0;
    for (int t4 = 0; t4 < TT / 2; t4 += 4) {
      unsigned int mv = *(const unsigned int*)(mrow + tbase + t4);
#pragma unroll
      for (int i = 0; i < 4; ++i)
        mLDS[(size_t)(tbase + t4 + i) * 16 + n] = (unsigned char)((mv >> (8 * i)) & 1u);
    }
  }
  // ---- seed q(0)[j] = exp(start[j]) (batch independent)
  for (int idx = l; idx < 16 * LL; idx += 32) {
    int nn = idx >> 6, j = idx & 63;
    qT[nn][j] = __expf(start_t[j]);
  }
  if (!hi) mLDS[n] = 1;   // reference applies t=0 unconditionally
  __syncthreads();

  // ---- constant A = exp(trans)^T as WMMA A operand (16x32 tiles), in this code's contiguous k order:
// lane row = l%16 = local j; element e -> local K = 16*(l/16) + e.
  v16h A[4][2];
#pragma unroll
  for (int jt = 0; jt < 4; ++jt)
#pragma unroll
    for (int kt = 0; kt < 2; ++kt) {
      v16h a;
#pragma unroll
      for (int e = 0; e < 16; ++e) {
        int d  = e >> 1;
        int kb = e + (hi ? 16 : 0); (void)d;
        int i  = kt * 32 + kb;       // source state (K)
        int j  = jt * 16 + n;        // dest state (row)
        a[e] = (_Float16)__expf(trans[i * LL + j]);
      }
      A[jt][kt] = a;
    }

  const float* hrow = h + (size_t)(b0 + n) * TT * LL;
  float logscale = 0.f;

  // ---- sequential scan over time
  for (int t = 1; t < TT; ++t) {
    bool mk = mLDS[(size_t)(t - 1) * 16 + n] != 0;

    // B-build: p(t-1) = mk ? q(t-1) * exp(h(t-1)) : p(t-2); track row max.
    float pv[32];
    float rmax = 0.f;
#pragma unroll
    for (int kt = 0; kt < 2; ++kt) {
      int jb = kt * 32 + koff;
      const float4* hq = (const float4*)(hrow + (size_t)(t - 1) * LL + jb);
      const float4* qq = (const float4*)(&qT[n][jb]);
      const float4* pq = (const float4*)(&pT[n][jb]);
#pragma unroll
      for (int c = 0; c < 4; ++c) {
        float4 qv = qq[c];
        float4 hv = hq[c];
        float4 p;
        if (mk) {
          p.x = qv.x * __expf(hv.x);
          p.y = qv.y * __expf(hv.y);
          p.z = qv.z * __expf(hv.z);
          p.w = qv.w * __expf(hv.w);
        } else {
          p = pq[c];
        }
        pv[kt * 16 + 4 * c + 0] = p.x;
        pv[kt * 16 + 4 * c + 1] = p.y;
        pv[kt * 16 + 4 * c + 2] = p.z;
        pv[kt * 16 + 4 * c + 3] = p.w;
        rmax = fmaxf(rmax, fmaxf(fmaxf(p.x, p.y), fmaxf(p.z, p.w)));
      }
    }
    // row max across the two half-wave K ranges of this batch
    rmax = fmaxf(rmax, __shfl_xor(rmax, 16, 32));
    rmax = fmaxf(rmax, 1e-30f);
    float rinv = 1.0f / rmax;
    logscale += __logf(rmax);           // masked rows have rmax==1 -> no-op

    // normalize, store pT, pack f16 B operand (lane = batch column)
    v16h Bv[2];
#pragma unroll
    for (int kt = 0; kt < 2; ++kt) {
      int jb = kt * 32 + koff;
      float4* pw = (float4*)(&pT[n][jb]);
#pragma unroll
      for (int c = 0; c < 4; ++c) {
        float4 pn;
        pn.x = pv[kt * 16 + 4 * c + 0] * rinv;
        pn.y = pv[kt * 16 + 4 * c + 1] * rinv;
        pn.z = pv[kt * 16 + 4 * c + 2] * rinv;
        pn.w = pv[kt * 16 + 4 * c + 3] * rinv;
        pw[c] = pn;
        Bv[kt][4 * c + 0] = (_Float16)pn.x;
        Bv[kt][4 * c + 1] = (_Float16)pn.y;
        Bv[kt][4 * c + 2] = (_Float16)pn.z;
        Bv[kt][4 * c + 3] = (_Float16)pn.w;
      }
    }

    // q(t) = expTrans^T @ p(t-1): 4 j-tiles x 2 K-chunks, K=32 each
#pragma unroll
    for (int jt = 0; jt < 4; ++jt) {
      v8f c2 = {};
      c2 = __builtin_amdgcn_wmma_f32_16x16x32_f16(false, A[jt][0], false, Bv[0],
                                                  (short)0, c2, false, false);
      c2 = __builtin_amdgcn_wmma_f32_16x16x32_f16(false, A[jt][1], false, Bv[1],
                                                  (short)0, c2, false, false);
      int jo = jt * 16 + (hi ? 8 : 0);
#pragma unroll
      for (int r = 0; r < 8; ++r) qT[n][jo + r] = c2[r];  // D scatter -> qT
    }

    // prefetch h a few steps ahead (global_prefetch_b8)
    if (t + 3 < TT)
      __builtin_prefetch(hrow + (size_t)(t + 3) * LL + (hi ? 32 : 0), 0, 1);
  }

  // ---- final: p(T-1), then logZ = logscale + log(sum_j p * exp(end))
  {
    bool mk = mLDS[(size_t)(TT - 1) * 16 + n] != 0;
    float sum = 0.f;
#pragma unroll
    for (int kt = 0; kt < 2; ++kt) {
      int jb = kt * 32 + koff;
      const float4* hq = (const float4*)(hrow + (size_t)(TT - 1) * LL + jb);
      const float4* qq = (const float4*)(&qT[n][jb]);
      const float4* pq = (const float4*)(&pT[n][jb]);
#pragma unroll
      for (int c = 0; c < 4; ++c) {
        float4 qv = qq[c];
        float4 hv = hq[c];
        float4 p;
        if (mk) {
          p.x = qv.x * __expf(hv.x);
          p.y = qv.y * __expf(hv.y);
          p.z = qv.z * __expf(hv.z);
          p.w = qv.w * __expf(hv.w);
        } else {
          p = pq[c];
        }
        int j = jb + 4 * c;
        sum += p.x * __expf(end_t[j + 0]);
        sum += p.y * __expf(end_t[j + 1]);
        sum += p.z * __expf(end_t[j + 2]);
        sum += p.w * __expf(end_t[j + 3]);
      }
    }
    sum += __shfl_xor(sum, 16, 32);
    float logZ = logscale + __logf(sum);
    if (!hi) out[b0 + n] -= logZ;   // out preloaded with numerator
  }
}

// ------------------------------------------------------------------ launch
extern "C" void kernel_launch(void* const* d_in, const int* in_sizes, int n_in,
                              void* d_out, int out_size, void* d_ws, size_t ws_size,
                              hipStream_t stream) {
  (void)in_sizes; (void)n_in; (void)out_size; (void)d_ws; (void)ws_size;
  const float*         h       = (const float*)d_in[0];
  const long long*     labels  = (const long long*)d_in[1];
  const unsigned char* mask    = (const unsigned char*)d_in[2];
  const float*         trans   = (const float*)d_in[3];
  const float*         start_t = (const float*)d_in[4];
  const float*         end_t   = (const float*)d_in[5];
  float*               out     = (float*)d_out;

  crf_num_kernel<<<BB, 256, 0, stream>>>(h, labels, mask, trans, start_t, end_t, out);
  crf_den_kernel<<<BB / 16, 32, 0, stream>>>(h, mask, trans, start_t, end_t, out);
}
